// GAT_69277822484757
// MI455X (gfx1250) — matrix operand order repaired and checked by reading, not compiled, not run on hardware
//
#include <hip/hip_runtime.h>
#include <hip/hip_bf16.h>

typedef __attribute__((ext_vector_type(16))) _Float16 v16h;
typedef __attribute__((ext_vector_type(8)))  float    v8f;

#define NEG_SLOPE 0.2f
#define GAT_EPS 1e-16f
#define NEG_HUGE -3.402823466e38f

// ---------------------------------------------------------------- utilities

__global__ void fill_f32(float* __restrict__ p, float v, long n) {
    long i = (long)blockIdx.x * blockDim.x + threadIdx.x;
    if (i < n) p[i] = v;
}

__device__ __forceinline__ void atomicMaxF(float* addr, float v) {
    // int compare is monotonic for non-negative floats; uint compare is
    // reverse-monotonic for negative floats. Init must be very negative.
    if (v >= 0.0f) atomicMax((int*)addr, __float_as_int(v));
    else           atomicMin((unsigned int*)addr, __float_as_uint(v));
}

__device__ __forceinline__ float lrelu(float x) {
    return x > 0.0f ? x : NEG_SLOPE * x;
}

// ---------------------------------------------------------------- GEMM 1
// h1[N,64] = x[N,128] @ W1[128,64], f16 WMMA, f32 accumulate.
// Block: 256 threads = 8 waves, each wave owns a 16-row tile (128 rows/block).
// W1 is staged in LDS pre-swizzled into WMMA B-fragment layout so each
// fragment is one contiguous 32B per-lane read (ds_load_b128 x2).
__global__ __launch_bounds__(256)
void gemm1_wmma(const float* __restrict__ x, const float* __restrict__ W,
                float* __restrict__ h1, int N) {
    // fragment layout: [frag(kt*4+nt)][lane][i] halves; 16 frags * 32 * 16
    __shared__ _Float16 sW[16 * 32 * 16];
    for (int idx = threadIdx.x; idx < 16 * 32 * 16; idx += 256) {
        int i    = idx & 15;
        int lane = (idx >> 4) & 31;
        int frag = idx >> 9;                 // 0..15
        int kt   = frag >> 2, nt = frag & 3;
        int k    = kt * 32 + ((i >> 3) * 16) + ((lane >> 4) * 8) + (i & 7);   // B frag: k = (i>>3)*16 + half*8 + (i&7)
        int n    = nt * 16 + (lane & 15);              // col = lane%16
        sW[idx] = (_Float16)W[k * 64 + n];
    }
    __syncthreads();

    const int wave    = threadIdx.x >> 5;
    const int lane    = threadIdx.x & 31;
    const int half_id = (lane >> 4) & 1;
    const int mr      = lane & 15;
    const int rowBase = blockIdx.x * 128 + wave * 16;

    int row = rowBase + mr;
    if (row >= N) row = N - 1;           // clamp loads; stores are guarded
    const float4* xv = (const float4*)x; // 128 f32 = 32 float4 per row
    const v16h* sWv = (const v16h*)sW;

    v8f acc[4];
    #pragma unroll
    for (int nt = 0; nt < 4; nt++) acc[nt] = (v8f){0,0,0,0,0,0,0,0};

    #pragma unroll
    for (int kt = 0; kt < 4; kt++) {
        const int k0 = kt * 32;
        // A fragment: element i -> k = k0 + (i>>3)*16 + half_id*8 + (i&7)
        const long rb = (long)row * 32;
        float4 fa = xv[rb + ((k0 +      half_id * 8) >> 2)];
        float4 fb = xv[rb + ((k0 +      half_id * 8) >> 2) + 1];
        float4 fc = xv[rb + ((k0 + 16 + half_id * 8) >> 2)];
        float4 fd = xv[rb + ((k0 + 16 + half_id * 8) >> 2) + 1];
        v16h a;
        a[0]=(_Float16)fa.x;  a[1]=(_Float16)fa.y;  a[2]=(_Float16)fa.z;  a[3]=(_Float16)fa.w;
        a[4]=(_Float16)fb.x;  a[5]=(_Float16)fb.y;  a[6]=(_Float16)fb.z;  a[7]=(_Float16)fb.w;
        a[8]=(_Float16)fc.x;  a[9]=(_Float16)fc.y;  a[10]=(_Float16)fc.z; a[11]=(_Float16)fc.w;
        a[12]=(_Float16)fd.x; a[13]=(_Float16)fd.y; a[14]=(_Float16)fd.z; a[15]=(_Float16)fd.w;

        #pragma unroll
        for (int nt = 0; nt < 4; nt++) {
            v16h b = sWv[(kt * 4 + nt) * 32 + lane];   // one 32B LDS read
            acc[nt] = __builtin_amdgcn_wmma_f32_16x16x32_f16(
                false, a, false, b, (short)0, acc[nt], false, false);
        }
    }

    #pragma unroll
    for (int nt = 0; nt < 4; nt++) {
        #pragma unroll
        for (int r = 0; r < 8; r++) {
            int grow = rowBase + r + half_id * 8;
            if (grow < N)
                h1[(long)grow * 64 + nt * 16 + mr] = acc[nt][r];
        }
    }
}

// ---------------------------------------------------------------- GEMM 2
// h2[N,16] = hin[N,64] @ W2[64,16]
__global__ __launch_bounds__(256)
void gemm2_wmma(const float* __restrict__ hin, const float* __restrict__ W,
                float* __restrict__ h2, int N) {
    // fragment layout: [frag(kt)][lane][i]; 2 frags * 32 * 16 halves
    __shared__ _Float16 sW[2 * 32 * 16];
    for (int idx = threadIdx.x; idx < 2 * 32 * 16; idx += 256) {
        int i    = idx & 15;
        int lane = (idx >> 4) & 31;
        int kt   = idx >> 9;
        int k    = kt * 32 + ((i >> 3) * 16) + ((lane >> 4) * 8) + (i & 7);
        int n    = lane & 15;
        sW[idx] = (_Float16)W[k * 16 + n];
    }
    __syncthreads();

    const int wave    = threadIdx.x >> 5;
    const int lane    = threadIdx.x & 31;
    const int half_id = (lane >> 4) & 1;
    const int mr      = lane & 15;
    const int rowBase = blockIdx.x * 128 + wave * 16;

    int row = rowBase + mr;
    if (row >= N) row = N - 1;
    const float4* xv = (const float4*)hin; // 64 f32 = 16 float4 per row
    const v16h* sWv = (const v16h*)sW;

    v8f acc = (v8f){0,0,0,0,0,0,0,0};
    #pragma unroll
    for (int kt = 0; kt < 2; kt++) {
        const int k0 = kt * 32;
        const long rb = (long)row * 16;
        float4 fa = xv[rb + ((k0 +      half_id * 8) >> 2)];
        float4 fb = xv[rb + ((k0 +      half_id * 8) >> 2) + 1];
        float4 fc = xv[rb + ((k0 + 16 + half_id * 8) >> 2)];
        float4 fd = xv[rb + ((k0 + 16 + half_id * 8) >> 2) + 1];
        v16h a;
        a[0]=(_Float16)fa.x;  a[1]=(_Float16)fa.y;  a[2]=(_Float16)fa.z;  a[3]=(_Float16)fa.w;
        a[4]=(_Float16)fb.x;  a[5]=(_Float16)fb.y;  a[6]=(_Float16)fb.z;  a[7]=(_Float16)fb.w;
        a[8]=(_Float16)fc.x;  a[9]=(_Float16)fc.y;  a[10]=(_Float16)fc.z; a[11]=(_Float16)fc.w;
        a[12]=(_Float16)fd.x; a[13]=(_Float16)fd.y; a[14]=(_Float16)fd.z; a[15]=(_Float16)fd.w;

        v16h b = sWv[kt * 32 + lane];
        acc = __builtin_amdgcn_wmma_f32_16x16x32_f16(
            false, a, false, b, (short)0, acc, false, false);
    }

    #pragma unroll
    for (int r = 0; r < 8; r++) {
        int grow = rowBase + r + half_id * 8;
        if (grow < N)
            h2[(long)grow * 16 + mr] = acc[r];
    }
}

// ---------------------------------------------------------------- attention scores
// a_src[n,h] = sum_c feat[n,h,c]*att_s[h,c]; a_dst likewise. C=16 fixed.
template <int H>
__global__ void att_scores(const float* __restrict__ feat,
                           const float* __restrict__ att_s,
                           const float* __restrict__ att_d,
                           float* __restrict__ as_, float* __restrict__ ad_,
                           int N) {
    int idx = blockIdx.x * blockDim.x + threadIdx.x;
    if (idx >= N * H) return;
    int n = idx / H, h = idx % H;
    const float* f = feat + (long)n * H * 16 + h * 16;
    float ds = 0.0f, dd = 0.0f;
    #pragma unroll
    for (int c = 0; c < 16; c++) {
        float v = f[c];
        ds += v * att_s[h * 16 + c];
        dd += v * att_d[h * 16 + c];
    }
    as_[idx] = ds;
    ad_[idx] = dd;
}

// ---------------------------------------------------------------- edge passes
// index t in [0, E+N): t < E -> real edge, else self loop (t-E, t-E)
template <int H>
__global__ void edge_max(const int* __restrict__ srcI, const int* __restrict__ dstI,
                         int E, int N,
                         const float* __restrict__ as_, const float* __restrict__ ad_,
                         float* __restrict__ m) {
    int t = blockIdx.x * blockDim.x + threadIdx.x;
    if (t >= E + N) return;
    int s = (t < E) ? srcI[t] : (t - E);
    int d = (t < E) ? dstI[t] : (t - E);
    #pragma unroll
    for (int h = 0; h < H; h++) {
        float e = lrelu(as_[s * H + h] + ad_[d * H + h]);
        atomicMaxF(&m[d * H + h], e);
    }
}

template <int H>
__global__ void edge_sum(const int* __restrict__ srcI, const int* __restrict__ dstI,
                         int E, int N,
                         const float* __restrict__ as_, const float* __restrict__ ad_,
                         const float* __restrict__ m, float* __restrict__ ssum) {
    int t = blockIdx.x * blockDim.x + threadIdx.x;
    if (t >= E + N) return;
    int s = (t < E) ? srcI[t] : (t - E);
    int d = (t < E) ? dstI[t] : (t - E);
    #pragma unroll
    for (int h = 0; h < H; h++) {
        float e = lrelu(as_[s * H + h] + ad_[d * H + h]);
        atomicAdd(&ssum[d * H + h], __expf(e - m[d * H + h]));
    }
}

template <int H>
__global__ void edge_aggr(const int* __restrict__ srcI, const int* __restrict__ dstI,
                          int E, int N,
                          const float* __restrict__ as_, const float* __restrict__ ad_,
                          const float* __restrict__ m, const float* __restrict__ ssum,
                          const float* __restrict__ feat, float* __restrict__ out) {
    int t = blockIdx.x * blockDim.x + threadIdx.x;
    if (t >= E + N) return;
    int s = (t < E) ? srcI[t] : (t - E);
    int d = (t < E) ? dstI[t] : (t - E);
    #pragma unroll
    for (int h = 0; h < H; h++) {
        float e = lrelu(as_[s * H + h] + ad_[d * H + h]);
        float alpha = __expf(e - m[d * H + h]) / (ssum[d * H + h] + GAT_EPS);
        const float4* hf = (const float4*)(feat + (long)s * H * 16 + h * 16);
        float* ob = out + (long)d * H * 16 + h * 16;
        #pragma unroll
        for (int c4 = 0; c4 < 4; c4++) {
            float4 q = hf[c4];
            atomicAdd(&ob[c4 * 4 + 0], q.x * alpha);
            atomicAdd(&ob[c4 * 4 + 1], q.y * alpha);
            atomicAdd(&ob[c4 * 4 + 2], q.z * alpha);
            atomicAdd(&ob[c4 * 4 + 3], q.w * alpha);
        }
    }
}

// ---------------------------------------------------------------- elementwise
__global__ void elu_bias(float* __restrict__ a, const float* __restrict__ bias, long n) {
    long i = (long)blockIdx.x * blockDim.x + threadIdx.x;
    if (i >= n) return;
    float v = a[i] + bias[i & 63];               // H1*C1 == 64
    a[i] = v > 0.0f ? v : __expf(v) - 1.0f;      // ELU
}

__global__ void final_logsoftmax(float* __restrict__ out, const float* __restrict__ bias2,
                                 int N) {
    int n = blockIdx.x * blockDim.x + threadIdx.x;
    if (n >= N) return;
    float v[16];
    float mx = NEG_HUGE;
    #pragma unroll
    for (int c = 0; c < 16; c++) {
        v[c] = out[(long)n * 16 + c] + bias2[c];
        mx = fmaxf(mx, v[c]);
    }
    float s = 0.0f;
    #pragma unroll
    for (int c = 0; c < 16; c++) s += __expf(v[c] - mx);
    float l = logf(s);
    #pragma unroll
    for (int c = 0; c < 16; c++) out[(long)n * 16 + c] = v[c] - mx - l;
}

// ---------------------------------------------------------------- host

extern "C" void kernel_launch(void* const* d_in, const int* in_sizes, int n_in,
                              void* d_out, int out_size, void* d_ws, size_t ws_size,
                              hipStream_t stream) {
    const float* x    = (const float*)d_in[0];
    const int*   ei   = (const int*)d_in[1];
    const float* W1   = (const float*)d_in[2];
    const float* atS1 = (const float*)d_in[3];
    const float* atD1 = (const float*)d_in[4];
    const float* b1   = (const float*)d_in[5];
    const float* W2   = (const float*)d_in[6];
    const float* atS2 = (const float*)d_in[7];
    const float* atD2 = (const float*)d_in[8];
    const float* b2   = (const float*)d_in[9];

    const int N = in_sizes[0] / 128;
    const int E = in_sizes[1] / 2;
    const int* srcI = ei;
    const int* dstI = ei + E;
    float* out = (float*)d_out;

    // workspace partition (all node arrays; ~66 MB for N=100k -> L2 resident)
    float* p = (float*)d_ws;
    float* h1   = p; p += (long)N * 64;
    float* as1  = p; p += (long)N * 4;
    float* ad1  = p; p += (long)N * 4;
    float* m1   = p; p += (long)N * 4;
    float* s1   = p; p += (long)N * 4;
    float* acc1 = p; p += (long)N * 64;   // becomes hin2 after ELU (in place)
    float* h2   = p; p += (long)N * 16;
    float* as2  = p; p += (long)N;
    float* ad2  = p; p += (long)N;
    float* m2   = p; p += (long)N;
    float* s2   = p; p += (long)N;

    const int B = 256;
    auto blocks = [](long n) { return (int)((n + 255) / 256); };
    const int T = E + N; // edges incl. self-loops

    // init accumulators / maxima / output
    fill_f32<<<blocks((long)N * 4), B, 0, stream>>>(m1, NEG_HUGE, (long)N * 4);
    fill_f32<<<blocks((long)N * 4), B, 0, stream>>>(s1, 0.0f, (long)N * 4);
    fill_f32<<<blocks((long)N * 64), B, 0, stream>>>(acc1, 0.0f, (long)N * 64);
    fill_f32<<<blocks(N), B, 0, stream>>>(m2, NEG_HUGE, N);
    fill_f32<<<blocks(N), B, 0, stream>>>(s2, 0.0f, N);
    fill_f32<<<blocks((long)N * 16), B, 0, stream>>>(out, 0.0f, (long)N * 16);

    // ----- layer 1 -----
    gemm1_wmma<<<(N + 127) / 128, B, 0, stream>>>(x, W1, h1, N);
    att_scores<4><<<blocks((long)N * 4), B, 0, stream>>>(h1, atS1, atD1, as1, ad1, N);
    edge_max<4><<<blocks(T), B, 0, stream>>>(srcI, dstI, E, N, as1, ad1, m1);
    edge_sum<4><<<blocks(T), B, 0, stream>>>(srcI, dstI, E, N, as1, ad1, m1, s1);
    edge_aggr<4><<<blocks(T), B, 0, stream>>>(srcI, dstI, E, N, as1, ad1, m1, s1, h1, acc1);
    elu_bias<<<blocks((long)N * 64), B, 0, stream>>>(acc1, b1, (long)N * 64);

    // ----- layer 2 -----
    gemm2_wmma<<<(N + 127) / 128, B, 0, stream>>>(acc1, W2, h2, N);
    att_scores<1><<<blocks(N), B, 0, stream>>>(h2, atS2, atD2, as2, ad2, N);
    edge_max<1><<<blocks(T), B, 0, stream>>>(srcI, dstI, E, N, as2, ad2, m2);
    edge_sum<1><<<blocks(T), B, 0, stream>>>(srcI, dstI, E, N, as2, ad2, m2, s2);
    edge_aggr<1><<<blocks(T), B, 0, stream>>>(srcI, dstI, E, N, as2, ad2, m2, s2, h2, out);

    final_logsoftmax<<<blocks(N), B, 0, stream>>>(out, b2, N);
}
